// DeformConv2d_90984587198679
// MI455X (gfx1250) — matrix operand order repaired and checked by reading, not compiled, not run on hardware
//
#include <hip/hip_runtime.h>

typedef __attribute__((ext_vector_type(16))) _Float16 v16h;
typedef __attribute__((ext_vector_type(8)))  _Float16 v8h;
typedef __attribute__((ext_vector_type(8)))  float    v8f;

#define B_    8
#define CIN   64
#define H_    128
#define W_    128
#define COUT  64
#define KTAP  9
#define KDIM  (CIN * KTAP)        // 576 = GEMM reduction dim
#define PIX_PER_WG 32
#define NPAIR (PIX_PER_WG * KTAP) // 288 (pixel, tap) pairs
#define COLSTRIDE (KDIM + 16)     // 592 f16 -> 1184 B row, 16B aligned
#define KCHUNKS (KDIM / 32)       // 18 WMMA K-steps

// ---------------------------------------------------------------------------
// Prep: weight (Cout,Cin,3,3) fp32 -> f16 matrix WA[co][kd], kd = tap*64 + c
// ---------------------------------------------------------------------------
__global__ void dcn_prep_weights(const float* __restrict__ w,
                                 _Float16* __restrict__ wa) {
    int idx = blockIdx.x * 256 + threadIdx.x;
    if (idx >= COUT * KDIM) return;
    int co  = idx / KDIM;
    int kd  = idx - co * KDIM;
    int tap = kd >> 6;            // kd / 64
    int c   = kd & 63;
    wa[co * KDIM + kd] = (_Float16)w[(co * CIN + c) * KTAP + tap];
}

// ---------------------------------------------------------------------------
// Fused sampling + WMMA GEMM. One workgroup = 32 consecutive output pixels.
// ---------------------------------------------------------------------------
struct Coef { float w00, w01, w10, w11; int yx0; int yx1; };

__launch_bounds__(256)
__global__ void dcn_main(const float* __restrict__ x,
                         const float* __restrict__ off,
                         const float* __restrict__ mask,
                         const _Float16* __restrict__ wa,
                         const float* __restrict__ bias,
                         float* __restrict__ out) {
    __shared__ __align__(16) _Float16 col[PIX_PER_WG * COLSTRIDE]; // 37888 B
    __shared__ Coef  cf[NPAIR];                                    //  6912 B
    __shared__ float sbias[COUT];

    const int tid  = threadIdx.x;
    const int wg   = blockIdx.x;
    const int b    = wg >> 9;           // 512 tiles per batch image
    const int tile = wg & 511;
    const int pix_base = tile * PIX_PER_WG;  // 32 pixels, same row (128%32==0)

    if (tid < COUT) sbias[tid] = bias[tid];

    // ---- Phase 1: bilinear coefficients for each (pixel, tap) pair --------
    for (int pp = tid; pp < NPAIR; pp += 256) {
        int pl  = pp / KTAP;
        int tap = pp - pl * KTAP;
        int pix = pix_base + pl;
        int ho  = pix >> 7;
        int wo  = pix & 127;
        int ky  = tap / 3, kx = tap - ky * 3;

        int   obase = ((b * 2 * KTAP + 2 * tap) << 14) + pix;
        float oy = off[obase];
        float ox = off[obase + (1 << 14)];
        float m  = mask[((b * KTAP + tap) << 14) + pix];

        float py = oy + (float)(ho + ky - 1);   // stride 1, pad 1, dil 1
        float px = ox + (float)(wo + kx - 1);
        float y0f = floorf(py), x0f = floorf(px);
        float ly = py - y0f,    lx = px - x0f;
        int y0 = (int)y0f, x0 = (int)x0f;
        int y1 = y0 + 1,   x1 = x0 + 1;
        bool vy0 = (y0 >= 0) & (y0 < H_);
        bool vy1 = (y1 >= 0) & (y1 < H_);
        bool vx0 = (x0 >= 0) & (x0 < W_);
        bool vx1 = (x1 >= 0) & (x1 < W_);
        int y0c = min(max(y0, 0), H_ - 1), y1c = min(max(y1, 0), H_ - 1);
        int x0c = min(max(x0, 0), W_ - 1), x1c = min(max(x1, 0), W_ - 1);

        Coef c;
        c.w00 = (vy0 && vx0) ? (1.f - ly) * (1.f - lx) * m : 0.f;
        c.w01 = (vy0 && vx1) ? (1.f - ly) * lx         * m : 0.f;
        c.w10 = (vy1 && vx0) ? ly         * (1.f - lx) * m : 0.f;
        c.w11 = (vy1 && vx1) ? ly         * lx         * m : 0.f;
        c.yx0 = (y0c << 16) | x0c;
        c.yx1 = (y1c << 16) | x1c;
        cf[pp] = c;
    }
    __syncthreads();

    // ---- Phase 2: gather 64 channels per pair -> col (f16) ----------------
    // i = c*NPAIR + pp: consecutive lanes walk consecutive (px,tap) within
    // one channel plane -> spatially local gathers.
    const float* xb = x + ((b * CIN) << 14);
    for (int i = tid; i < NPAIR * CIN; i += 256) {
        int c  = i / NPAIR;
        int pp = i - c * NPAIR;
        Coef cc = cf[pp];
        int y0 = cc.yx0 >> 16, x0 = cc.yx0 & 0xffff;
        int y1 = cc.yx1 >> 16, x1 = cc.yx1 & 0xffff;
        const float* plane = xb + (c << 14);
        float v = cc.w00 * plane[(y0 << 7) + x0]
                + cc.w01 * plane[(y0 << 7) + x1]
                + cc.w10 * plane[(y1 << 7) + x0]
                + cc.w11 * plane[(y1 << 7) + x1];
        int pl  = pp / KTAP;
        int tap = pp - pl * KTAP;
        col[pl * COLSTRIDE + (tap << 6) + c] = (_Float16)v;
    }
    __syncthreads();

    // ---- Phase 3: WMMA. 8 waves: 4 M-tiles (Cout) x 2 N-tiles (pixels) ----
    const int wave = tid >> 5;
    const int lane = tid & 31;
    const int mt   = wave & 3;
    const int nt   = wave >> 2;
    const int lrow = lane & 15;
    const int half = lane >> 4;

    // A 16x32 f16 layout: lane row = lane&15; halves hold K {0-7,16-23} /
    // {8-15,24-31}. B 32x16 f16: lane col = lane&15; same K split as A.
    const _Float16* warow  = wa  + (size_t)(mt * 16 + lrow) * KDIM + half * 8;
    const _Float16* colrow = col + (nt * 16 + lrow) * COLSTRIDE + half * 8;

    v8f acc = {};
    for (int kc = 0; kc < KCHUNKS; ++kc) {
        int kb = kc * 32;
        union { v16h v; v8h h[2]; } A, Bm;
        A.h[0]  = *(const v8h*)(warow  + kb);        // K kb+half*8 .. +7
        A.h[1]  = *(const v8h*)(warow  + kb + 16);   // K kb+half*8+16 .. +23
        Bm.h[0] = *(const v8h*)(colrow + kb);        // K kb+half*8 .. +7
        Bm.h[1] = *(const v8h*)(colrow + kb + 16);   // K kb+half*8+16 .. +23
        acc = __builtin_amdgcn_wmma_f32_16x16x32_f16(
                  false, A.v, false, Bm.v, (short)0, acc, false, false);
    }

    // ---- Phase 4: store D (+bias). C/D layout: VGPR r, lanes<16 -> M=r,
    // lanes>=16 -> M=8+r; N = lane&15. ----------------------------------
    int pxg = pix_base + nt * 16 + lrow;
    float* outb = out + ((b * COUT) << 14) + pxg;
    #pragma unroll
    for (int r = 0; r < 8; ++r) {
        int co = mt * 16 + half * 8 + r;
        outb[co << 14] = acc[r] + sbias[co];
    }
}

// ---------------------------------------------------------------------------
extern "C" void kernel_launch(void* const* d_in, const int* in_sizes, int n_in,
                              void* d_out, int out_size, void* d_ws, size_t ws_size,
                              hipStream_t stream) {
    const float* x    = (const float*)d_in[0];   // (8,64,128,128)
    const float* off  = (const float*)d_in[1];   // (8,18,128,128)
    const float* mask = (const float*)d_in[2];   // (8,9,128,128)
    const float* wgt  = (const float*)d_in[3];   // (64,64,3,3)
    const float* bias = (const float*)d_in[4];   // (64,)
    float* out = (float*)d_out;                  // (8,64,128,128)

    _Float16* wa = (_Float16*)d_ws;              // 64*576 f16 = 73728 B

    int nprep = COUT * KDIM;
    dcn_prep_weights<<<(nprep + 255) / 256, 256, 0, stream>>>(wgt, wa);

    int ngrid = B_ * ((H_ * W_) / PIX_PER_WG);   // 8 * 512 = 4096 workgroups
    dcn_main<<<ngrid, 256, 0, stream>>>(x, off, mask, wa, bias, out);
}
